// PitchVQVAE_85263690760675
// MI455X (gfx1250) — matrix operand order repaired and checked by reading, not compiled, not run on hardware
//
#include <hip/hip_runtime.h>
#include <hip/hip_bf16.h>

typedef __attribute__((ext_vector_type(16))) _Float16 v16h;
typedef __attribute__((ext_vector_type(8)))  _Float16 h8;
typedef __attribute__((ext_vector_type(8)))  float    v8f;

// Problem constants
#define BB  32
#define SS  2048
#define DD  128
#define HH  256
#define KK  512
#define NN  (BB*SS)   // 65536 positions

// ---- workspace layout (bytes) ----
#define OFF_BUFA 0u                                  // f16 [B][256][S]  33,554,432
#define OFF_BUFB (OFF_BUFA + 33554432u)              // f16 [B][256][S]  33,554,432
#define OFF_ZT   (OFF_BUFB + 33554432u)              // f16 [N][128]     16,777,216
#define OFF_ZQ   (OFF_ZT   + 16777216u)              // f16 [B][128][S]  16,777,216
#define OFF_WE2  (OFF_ZQ   + 16777216u)              // f16 [256][3][256]   393,216
#define OFF_WE3  (OFF_WE2  + 393216u)                // f16 [128][3][256]   196,608
#define OFF_WD1  (OFF_WE3  + 196608u)                // f16 [256][3][128]   196,608
#define OFF_WD2  (OFF_WD1  + 196608u)                // f16 [256][3][256]   393,216
#define OFF_CBH  (OFF_WD2  + 393216u)                // f16 [512][128]      131,072
#define OFF_Z2   (OFF_CBH  + 131072u)                // f32 [N]             262,144
#define OFF_C2   (OFF_Z2   + 262144u)                // f32 [512]             2,048
#define OFF_ACC  (OFF_C2   + 2048u)                  // f32 [4]

// ---------------- prep kernels ----------------

__global__ void zero_acc_k(float* a) {
    if (threadIdx.x < 4) a[threadIdx.x] = 0.0f;
}

// src: f32 [co][ci][3]  ->  dst: f16 [co][t][ci]
__global__ void cvt_w_k(const float* __restrict__ src, _Float16* __restrict__ dst,
                        int CO, int CI) {
    int idx = blockIdx.x * 256 + threadIdx.x;
    int tot = CO * CI * 3;
    if (idx >= tot) return;
    int co  = idx / (CI * 3);
    int rem = idx - co * CI * 3;
    int ci  = rem / 3;
    int t   = rem - ci * 3;
    dst[(size_t)co * 3 * CI + (size_t)t * CI + ci] = (_Float16)src[idx];
}

// codebook f32 [512][128] -> f16 copy + per-row squared norms
__global__ void cb_prep_k(const float* __restrict__ cb, _Float16* __restrict__ cbh,
                          float* __restrict__ c2) {
    int k = blockIdx.x * 64 + threadIdx.x;
    if (k >= KK) return;
    float s = 0.0f;
    for (int d = 0; d < DD; ++d) {
        float v = cb[k * DD + d];
        cbh[k * DD + d] = (_Float16)v;
        s += v * v;
    }
    c2[k] = s;
}

// ---------------- encoder conv1 (1 -> 256), VALU (K=3 too small for WMMA) ----------------
__global__ __launch_bounds__(256)
void enc_conv1_k(const float* __restrict__ x, const float* __restrict__ w /*[256][1][3]*/,
                 const float* __restrict__ bias, _Float16* __restrict__ y /*[B][256][S]*/) {
    int idx = blockIdx.x * 256 + threadIdx.x;      // b*H*S + co*S + s
    int s  = idx & (SS - 1);
    int co = (idx >> 11) & (HH - 1);
    int b  = idx >> 19;
    const float* xb = x + (size_t)b * SS;
    float acc = bias[co];
    float w0 = w[co * 3 + 0], w1 = w[co * 3 + 1], w2 = w[co * 3 + 2];
    if (s > 0)      acc += w0 * xb[s - 1];
    acc += w1 * xb[s];
    if (s < SS - 1) acc += w2 * xb[s + 1];
    acc = (acc >= 0.0f) ? acc : 0.2f * acc;
    y[idx] = (_Float16)acc;
}

// ---------------- generic k=3 "same" conv1d as 3 shifted GEMMs via WMMA ----------------
// xin : f16 [B][CIN][S]
// wf  : f16 [COUT][3][CIN]
// LDS tile is stored POSITION-MAJOR so each lane's B fragment (K = half*8+0..7, 16+half*8+0..7)
// is two runs of 8 contiguous halves -> 2x ds_load_b128 instead of 16x ds_load_u16.
// grid: (S/16, COUT/64, B), block 128 (4 waves; each wave owns a 16-co tile)
#define LPAD 8
__global__ __launch_bounds__(128)
void conv1d_wmma_f16(const _Float16* __restrict__ xin,
                     const _Float16* __restrict__ wf,
                     const float*    __restrict__ bias,
                     _Float16*       __restrict__ yout,
                     int CIN, int COUT, int doLrelu, int transposeOut) {
    __shared__ __attribute__((aligned(32))) _Float16 lx[18 * (256 + LPAD)]; // max CIN=256
    const int LROW  = CIN + LPAD;                    // padded row stride (banks 4 apart)
    const int lane  = threadIdx.x & 31;
    const int wid   = threadIdx.x >> 5;
    const int half  = lane >> 4;
    const int l16   = lane & 15;
    const int sBase = blockIdx.x * 16;
    const int coB   = blockIdx.y * 64 + wid * 16;
    const int b     = blockIdx.z;

    // ---- stage the full 18 x CIN activation tile once ----
    const _Float16* xb = xin + (size_t)b * CIN * SS;
    const int total = 18 * CIN;
    for (int e = threadIdx.x; e < total; e += 128) {
        int ci = e / 18, so = e - ci * 18;           // so fastest across threads -> coalesced
        int s = sBase + so - 1;
        _Float16 v = (_Float16)0.0f;
        if (s >= 0 && s < SS) v = xb[(size_t)ci * SS + s];
        lx[so * LROW + ci] = v;
    }
    __syncthreads();

    v8f acc = {};
    const _Float16* wrow = wf + (size_t)(coB + l16) * 3 * CIN;

    for (int c0 = 0; c0 < CIN; c0 += 32) {
        #pragma unroll
        for (int t = 0; t < 3; ++t) {
            // A fragment (weights): K = half*8+0..7 and 16+half*8+0..7 -> two h8 loads
            const _Float16* wt = wrow + t * CIN + c0;
            h8 alo = *(const h8*)(wt + half * 8);
            h8 ahi = *(const h8*)(wt + 16 + half * 8);
            v16h a;
            #pragma unroll
            for (int i = 0; i < 8; ++i) { a[i] = alo[i]; a[8 + i] = ahi[i]; }

            // B fragment (activations): K = half*8+0..7 and 16+half*8+0..7 -> 2x ds_load_b128
            const _Float16* bp = lx + (l16 + t) * LROW + c0 + half * 8;
            h8 blo = *(const h8*)(bp);
            h8 bhi = *(const h8*)(bp + 16);
            v16h bf;
            #pragma unroll
            for (int i = 0; i < 8; ++i) { bf[i] = blo[i]; bf[8 + i] = bhi[i]; }

            acc = __builtin_amdgcn_wmma_f32_16x16x32_f16(false, a, false, bf,
                                                         (short)0, acc, false, false);
        }
    }

    // epilogue: C/D layout -> element r is (M = half*8 + r, N = l16)
    int s = sBase + l16;
    #pragma unroll
    for (int r = 0; r < 8; ++r) {
        int co = coB + half * 8 + r;
        float v = acc[r] + bias[co];
        if (doLrelu) v = (v >= 0.0f) ? v : 0.2f * v;
        if (!transposeOut)
            yout[((size_t)b * COUT + co) * SS + s] = (_Float16)v;
        else  // z_flat[n][d] for the VQ GEMM
            yout[(size_t)(b * SS + s) * COUT + co] = (_Float16)v;
    }
}

// ---------------- per-position squared norms of z ----------------
__global__ __launch_bounds__(256)
void z2_k(const _Float16* __restrict__ zt, float* __restrict__ z2) {
    int n = blockIdx.x * 256 + threadIdx.x;
    if (n >= NN) return;
    const _Float16* r = zt + (size_t)n * DD;
    float s = 0.0f;
    for (int d = 0; d < DD; d += 8) {
        h8 v = *(const h8*)(r + d);
        #pragma unroll
        for (int j = 0; j < 8; ++j) { float f = (float)v[j]; s += f * f; }
    }
    z2[n] = s;
}

// ---------------- VQ nearest-neighbor via WMMA + argmin reduction ----------------
// each wave handles 16 positions vs all 512 codes: 32 code tiles x 4 K-chunks = 128 WMMA
__global__ __launch_bounds__(128)
void vq_k(const _Float16* __restrict__ zt,  // [N][128]
          const _Float16* __restrict__ cbh, // [512][128]
          const float* __restrict__ z2, const float* __restrict__ c2,
          _Float16* __restrict__ zq,        // [B][128][S]
          float* __restrict__ outIdx,       // d_out + 65536 (indices as float)
          float* __restrict__ accum) {
    const int lane = threadIdx.x & 31;
    const int wid  = threadIdx.x >> 5;
    const int half = lane >> 4, l16 = lane & 15;
    const int nBase = blockIdx.x * 64 + wid * 16;

    // A fragments for all 4 K-chunks (z rows are contiguous in d)
    v16h A[4];
    const _Float16* zrow = zt + (size_t)(nBase + l16) * DD;
    #pragma unroll
    for (int ch = 0; ch < 4; ++ch) {
        h8 alo = *(const h8*)(zrow + ch * 32 + half * 16);
        h8 ahi = *(const h8*)(zrow + ch * 32 + half * 16 + 8);
        #pragma unroll
        for (int i = 0; i < 8; ++i) { A[ch][i] = alo[i]; A[ch][8 + i] = ahi[i]; }
    }
    float z2r[8];
    #pragma unroll
    for (int r = 0; r < 8; ++r) z2r[r] = z2[nBase + half * 8 + r];

    float minv[8]; int mini[8];
    #pragma unroll
    for (int r = 0; r < 8; ++r) { minv[r] = 3.4e38f; mini[r] = 0; }

    for (int kt = 0; kt < 32; ++kt) {
        int kBase = kt * 16;
        v8f acc = {};
        #pragma unroll
        for (int ch = 0; ch < 4; ++ch) {
            // B fragment: 16 contiguous halves at 32-B alignment -> single v16h load
            v16h bf = *(const v16h*)(cbh + (size_t)(kBase + l16) * DD + ch * 32 + half * 16);
            acc = __builtin_amdgcn_wmma_f32_16x16x32_f16(false, A[ch], false, bf,
                                                         (short)0, acc, false, false);
        }
        float c2k = c2[kBase + l16];
        #pragma unroll
        for (int r = 0; r < 8; ++r) {
            float dist = z2r[r] + c2k - 2.0f * acc[r];
            if (dist < minv[r]) { minv[r] = dist; mini[r] = kBase + l16; }
        }
    }

    // argmin across the 16 lanes of each half (tie-break: lowest code index)
    #pragma unroll
    for (int r = 0; r < 8; ++r) {
        #pragma unroll
        for (int off = 1; off < 16; off <<= 1) {
            float ov = __shfl_xor(minv[r], off, 32);
            int   oi = __shfl_xor(mini[r], off, 32);
            if (ov < minv[r] || (ov == minv[r] && oi < mini[r])) {
                minv[r] = ov; mini[r] = oi;
            }
        }
    }

    if (l16 == 0) {
        float sumd = 0.0f;
        #pragma unroll
        for (int r = 0; r < 8; ++r) {
            int p = nBase + half * 8 + r;
            outIdx[p] = (float)mini[r];
            sumd += fmaxf(minv[r], 0.0f);   // ||z - z_q||^2 (vq+commit numerator)
        }
        atomicAdd(accum + 0, sumd);
    }

    // scatter z_q = codebook[idx] into decoder layout [b][d][s] (all 16 lanes per half)
    #pragma unroll
    for (int r = 0; r < 8; ++r) {
        int p = nBase + half * 8 + r;
        int bb = p >> 11, s = p & (SS - 1);
        int kwin = mini[r];
        const _Float16* cr = cbh + (size_t)kwin * DD + l16 * 8;
        _Float16* zr = zq + ((size_t)bb * DD + l16 * 8) * SS + s;
        #pragma unroll
        for (int j = 0; j < 8; ++j) zr[(size_t)j * SS] = cr[j];
    }
}

// ---------------- decoder conv3 (256 -> 1) + recon loss ----------------
__global__ __launch_bounds__(256)
void dec_conv3_k(const _Float16* __restrict__ h /*[B][256][S]*/,
                 const float* __restrict__ w /*[256][3]*/, const float* __restrict__ bias,
                 const float* __restrict__ x, float* __restrict__ xrec,
                 float* __restrict__ accum) {
    int idx = blockIdx.x * 256 + threadIdx.x;     // b*S + s
    int s = idx & (SS - 1), b = idx >> 11;
    const _Float16* hb = h + (size_t)b * HH * SS;
    float acc = bias[0];
    for (int ci = 0; ci < HH; ++ci) {
        const _Float16* hr = hb + (size_t)ci * SS + s;
        float w0 = w[ci * 3 + 0], w1 = w[ci * 3 + 1], w2 = w[ci * 3 + 2];
        if (s > 0)      acc += w0 * (float)hr[-1];
        acc += w1 * (float)hr[0];
        if (s < SS - 1) acc += w2 * (float)hr[1];
    }
    xrec[idx] = acc;
    float d = x[idx] - acc;
    float sq = d * d;
    #pragma unroll
    for (int off = 16; off > 0; off >>= 1) sq += __shfl_xor(sq, off, 32);
    if ((threadIdx.x & 31) == 0) atomicAdd(accum + 1, sq);
}

__global__ void finalize_k(const float* __restrict__ a, float* __restrict__ out) {
    float recon = a[1] / (float)NN;                       // * RECON_COEF(1.0)
    float m     = a[0] / ((float)NN * (float)DD);
    float vq     = m * 0.05f;
    float commit = m * 0.15f;
    out[0] = recon + vq + commit;
    out[1] = recon;
    out[2] = vq;
    out[3] = commit;
}

// ---------------- launch ----------------
extern "C" void kernel_launch(void* const* d_in, const int* in_sizes, int n_in,
                              void* d_out, int out_size, void* d_ws, size_t ws_size,
                              hipStream_t stream) {
    (void)in_sizes; (void)n_in; (void)out_size; (void)ws_size;
    const float* x   = (const float*)d_in[0];
    const float* cb  = (const float*)d_in[1];
    const float* ew1 = (const float*)d_in[2];
    const float* eb1 = (const float*)d_in[3];
    const float* ew2 = (const float*)d_in[4];
    const float* eb2 = (const float*)d_in[5];
    const float* ew3 = (const float*)d_in[6];
    const float* eb3 = (const float*)d_in[7];
    const float* dw1 = (const float*)d_in[8];
    const float* db1 = (const float*)d_in[9];
    const float* dw2 = (const float*)d_in[10];
    const float* db2 = (const float*)d_in[11];
    const float* dw3 = (const float*)d_in[12];
    const float* db3 = (const float*)d_in[13];
    float* out = (float*)d_out;

    char* ws = (char*)d_ws;
    _Float16* bufA = (_Float16*)(ws + OFF_BUFA);
    _Float16* bufB = (_Float16*)(ws + OFF_BUFB);
    _Float16* zt   = (_Float16*)(ws + OFF_ZT);
    _Float16* zq   = (_Float16*)(ws + OFF_ZQ);
    _Float16* wfe2 = (_Float16*)(ws + OFF_WE2);
    _Float16* wfe3 = (_Float16*)(ws + OFF_WE3);
    _Float16* wfd1 = (_Float16*)(ws + OFF_WD1);
    _Float16* wfd2 = (_Float16*)(ws + OFF_WD2);
    _Float16* cbh  = (_Float16*)(ws + OFF_CBH);
    float*    z2   = (float*)(ws + OFF_Z2);
    float*    c2   = (float*)(ws + OFF_C2);
    float*    acc  = (float*)(ws + OFF_ACC);

    // prep (ws is poisoned before timing -> rebuild every launch; deterministic)
    zero_acc_k<<<1, 32, 0, stream>>>(acc);
    cvt_w_k<<<(HH * HH * 3 + 255) / 256, 256, 0, stream>>>(ew2, wfe2, HH, HH);
    cvt_w_k<<<(DD * HH * 3 + 255) / 256, 256, 0, stream>>>(ew3, wfe3, DD, HH);
    cvt_w_k<<<(HH * DD * 3 + 255) / 256, 256, 0, stream>>>(dw1, wfd1, HH, DD);
    cvt_w_k<<<(HH * HH * 3 + 255) / 256, 256, 0, stream>>>(dw2, wfd2, HH, HH);
    cb_prep_k<<<8, 64, 0, stream>>>(cb, cbh, c2);

    // encoder
    enc_conv1_k<<<(BB * HH * SS) / 256, 256, 0, stream>>>(x, ew1, eb1, bufA);
    conv1d_wmma_f16<<<dim3(SS / 16, HH / 64, BB), 128, 0, stream>>>(bufA, wfe2, eb2, bufB, HH, HH, 1, 0);
    conv1d_wmma_f16<<<dim3(SS / 16, DD / 64, BB), 128, 0, stream>>>(bufB, wfe3, eb3, zt, HH, DD, 0, 1);

    // vector quantization
    z2_k<<<NN / 256, 256, 0, stream>>>(zt, z2);
    vq_k<<<NN / 64, 128, 0, stream>>>(zt, cbh, z2, c2, zq, out + NN, acc);

    // decoder
    conv1d_wmma_f16<<<dim3(SS / 16, HH / 64, BB), 128, 0, stream>>>(zq, wfd1, db1, bufA, DD, HH, 1, 0);
    conv1d_wmma_f16<<<dim3(SS / 16, HH / 64, BB), 128, 0, stream>>>(bufA, wfd2, db2, bufB, HH, HH, 1, 0);
    dec_conv3_k<<<NN / 256, 256, 0, stream>>>(bufB, dw3, db3, x, out, acc);

    finalize_k<<<1, 1, 0, stream>>>(acc, out + 2 * NN);
}
